// Model_6914897346628
// MI455X (gfx1250) — compile-verified, matrix operand order checked by reading, not run on hardware
//
#include <hip/hip_runtime.h>
#include <cstddef>

// ---------------- WMMA types (gfx1250, wave32) -----------------------------
typedef __attribute__((ext_vector_type(16))) _Float16 v16h;
typedef __attribute__((ext_vector_type(8)))  float    v8f;
typedef __attribute__((ext_vector_type(4)))  int      v4i;

#if __has_builtin(__builtin_amdgcn_global_load_async_to_lds_b128) && \
    __has_builtin(__builtin_amdgcn_s_wait_asynccnt)
#define HAVE_ASYNC_LDS 1
#else
#define HAVE_ASYNC_LDS 0
#endif

// Fragment-element -> K offset within a 32-wide K tile (ISA 7.12.2, 16-bit A):
// lanes 0-15: kbase=0, lanes 16-31: kbase=8; vgpr v half h: v<4 -> K=2v+h,
// v>=4 -> K=16+2(v-4)+h.  Both A and B sides of every tile are produced and
// consumed with this same mapping (A: m=lane&15, B: n=lane&15).
__device__ __host__ static inline int frag_kk(int e) {
  const int v = e >> 1, h = e & 1;
  return (v < 4) ? (v * 2 + h) : 16 + ((v - 4) * 2 + h);
}

__device__ static inline v8f wmma_f16(v16h a, v16h b, v8f c) {
  return __builtin_amdgcn_wmma_f32_16x16x32_f16(false, a, false, b,
                                                (short)0, c, false, false);
}

// ---------------- Utility kernels ------------------------------------------
__global__ __launch_bounds__(256) void zero_f32_kernel(float* __restrict__ p, size_t n) {
  size_t i = (size_t)blockIdx.x * 256u + threadIdx.x;
  if (i < n) p[i] = 0.0f;
}

// Convert f32 weight [K x 128] -> f16 fragment-major [Kpad/32][8][32][16];
// rows K..Kpad-1 are zero.  Output index i = ((kt*8 + nt)*32 + lane)*16 + e.
__global__ __launch_bounds__(256) void cvt_frag_kernel(const float* __restrict__ in,
                                                       _Float16* __restrict__ out,
                                                       int K, int Kpad) {
  const int i = blockIdx.x * 256 + threadIdx.x;
  if (i >= Kpad * 128) return;
  const int e    = i & 15;
  const int lane = (i >> 4) & 31;
  const int nt   = (i >> 9) & 7;
  const int kt   = i >> 12;
  const int n  = nt * 16 + (lane & 15);
  const int kb = (lane & 16) ? 8 : 0;
  const int k  = kt * 32 + kb + frag_kk(e);
  out[i] = (_Float16)((k < K) ? in[(size_t)k * 128 + n] : 0.0f);
}

// Same, but K=256 formed by stacking two 128x128 weights [inA ; inB].
__global__ __launch_bounds__(256) void cvt_frag2_kernel(const float* __restrict__ inA,
                                                        const float* __restrict__ inB,
                                                        _Float16* __restrict__ out) {
  const int i = blockIdx.x * 256 + threadIdx.x;
  if (i >= 256 * 128) return;
  const int e    = i & 15;
  const int lane = (i >> 4) & 31;
  const int nt   = (i >> 9) & 7;
  const int kt   = i >> 12;
  const int n  = nt * 16 + (lane & 15);
  const int kb = (lane & 16) ? 8 : 0;
  const int k  = kt * 32 + kb + frag_kk(e);
  const float* src = (k < 128) ? inA : inB;
  out[i] = (_Float16)src[(size_t)(k & 127) * 128 + n];
}

// ---------------- Encoder: h = feat@encW + encB + PE(time)@timeW + timeB ----
// A tile staged in LDS in fragment order: As[(kt*32 + lane)*16 + e].
__global__ __launch_bounds__(256) void encode_kernel(
    const float* __restrict__ feat, int F,            // N x F (F=300)
    const float* __restrict__ t_node,
    const float* __restrict__ seed_time,
    const int*   __restrict__ batch,
    const _Float16* __restrict__ encW,                // frag-major, Kpad=320
    const float* __restrict__ encB,
    const _Float16* __restrict__ timeW,               // frag-major, Kpad=128
    const float* __restrict__ timeB,
    _Float16* __restrict__ hout)                      // N x 128 f16
{
  __shared__ __align__(32) _Float16 As[10 * 512];     // 10 K-tiles x 32 lanes x 16
  __shared__ float relS[16];
  const int tid  = threadIdx.x;
  const int lane = tid & 31;
  const int wave = tid >> 5;                          // 8 waves -> 8 N tiles
  const int row0 = blockIdx.x * 16;

  // Stage A1: features, fragment order, zero-pad K >= F.
  for (int idx = tid; idx < 10 * 512; idx += 256) {
    const int e  = idx & 15;
    const int l  = (idx >> 4) & 31;
    const int kt = idx >> 9;
    const int m  = l & 15;
    const int kb = (l & 16) ? 8 : 0;
    const int k  = kt * 32 + kb + frag_kk(e);
    const float v = (k < F) ? feat[(size_t)(row0 + m) * F + k] : 0.0f;
    As[idx] = (_Float16)v;
  }
  if (tid < 16) {
    const int r = row0 + tid;
    relS[tid] = (seed_time[batch[r]] - t_node[r]) * (1.0f / 86400.0f);
  }
  __syncthreads();

  v8f acc = {};
#pragma unroll
  for (int kt = 0; kt < 10; ++kt) {
    const v16h a = *(const v16h*)(As + ((size_t)kt * 32 + lane) * 16);
    const v16h b = *(const v16h*)(encW + (((size_t)kt * 8 + wave) * 32 + lane) * 16);
    acc = wmma_f16(a, b, acc);
  }
  __syncthreads();

  // Stage A2: positional encoding [sin(rel*f_j) | cos(rel*f_j)], fragment order.
  const float lstep = 0.14384103622589045f;           // ln(10000)/64
  for (int idx = tid; idx < 4 * 512; idx += 256) {
    const int e  = idx & 15;
    const int l  = (idx >> 4) & 31;
    const int kt = idx >> 9;
    const int m  = l & 15;
    const int kb = (l & 16) ? 8 : 0;
    const int c  = kt * 32 + kb + frag_kk(e);         // 0..127
    const int j  = c & 63;
    const float fr = __expf(-(float)j * lstep);
    const float ph = relS[m] * fr;
    As[idx] = (_Float16)((c < 64) ? __sinf(ph) : __cosf(ph));
  }
  __syncthreads();
#pragma unroll
  for (int kt = 0; kt < 4; ++kt) {
    const v16h a = *(const v16h*)(As + ((size_t)kt * 32 + lane) * 16);
    const v16h b = *(const v16h*)(timeW + (((size_t)kt * 8 + wave) * 32 + lane) * 16);
    acc = wmma_f16(a, b, acc);
  }

  // Epilogue: C/D layout N=lane&15, M=vgpr + 8*(lane>=16).
  const int n     = wave * 16 + (lane & 15);
  const int rbase = (lane & 16) ? 8 : 0;
  const float bias = encB[n] + timeB[n];
#pragma unroll
  for (int j = 0; j < 8; ++j)
    hout[(size_t)(row0 + rbase + j) * 128 + n] = (_Float16)(acc[j] + bias);
}

// ---------------- Edge scatter: acc[dst] += h[src]; cnt[dst] += 1 ----------
__global__ __launch_bounds__(256) void scatter_kernel(
    const int* __restrict__ edges,   // [2][E] (row0 = src, row1 = dst)
    int E,
    const _Float16* __restrict__ hsrc,
    float* __restrict__ acc,
    float* __restrict__ cnt)
{
  const long gid = (long)blockIdx.x * 256 + threadIdx.x;
  const long e = gid >> 5;                 // 32 threads per edge, 4 channels each
  if (e >= E) return;
  const int g   = (int)(gid & 31);
  const int src = edges[e];
  const int dst = edges[(size_t)E + e];
  if (g == 0) atomicAdd(&cnt[dst], 1.0f);
  const int c0 = g * 4;
  const _Float16* hp = hsrc + (size_t)src * 128 + c0;
  float* ap = acc + (size_t)dst * 128 + c0;
  __builtin_prefetch(hp, 0, 0);            // global_prefetch_b8 (L2-resident table)
#pragma unroll
  for (int q = 0; q < 4; ++q) atomicAdd(ap + q, (float)hp[q]);
}

// ---------------- SAGE layer: h = relu(LN([m|x]@[Wmsg;Wself] + b)) ----------
__global__ __launch_bounds__(256) void sage_update_kernel(
    const float* __restrict__ acc_m,            // N x 128 (segment sum)
    const float* __restrict__ cnt,              // N
    const _Float16* __restrict__ wcat,          // frag-major [Wmsg;Wself], Kpad=256
    const float* __restrict__ bias,
    const float* __restrict__ ln_g,
    const float* __restrict__ ln_b,
    _Float16* __restrict__ h)                   // N x 128, updated in place
{
  __shared__ __align__(32) _Float16 As[8 * 512];
  __shared__ float Os[16 * 128];
  __shared__ float mu[16], rsig[16];
  const int tid  = threadIdx.x;
  const int lane = tid & 31;
  const int wave = tid >> 5;
  const int row0 = blockIdx.x * 16;

  // Stage A low half (k < 128): mean-message, needs f32 math -> VALU staging.
  for (int idx = tid; idx < 4 * 512; idx += 256) {
    const int e  = idx & 15;
    const int l  = (idx >> 4) & 31;
    const int kt = idx >> 9;                          // 0..3
    const int m  = l & 15;
    const int kb = (l & 16) ? 8 : 0;
    const int k  = kt * 32 + kb + frag_kk(e);         // 0..127
    const float c = cnt[row0 + m];
    const float v = acc_m[(size_t)(row0 + m) * 128 + k] / fmaxf(c, 1.0f);
    As[idx] = (_Float16)v;
  }

  // Stage A high half (k in [128,256)): raw f16 row copy of the L2-resident
  // node table.  Fragment layout makes each lane's share two contiguous 16-B
  // runs -> one GLOBAL_LOAD_ASYNC_TO_LDS_B128 per thread (ASYNCcnt-tracked).
#if HAVE_ASYNC_LDS
  {
    const int half = tid & 1;                         // 8-element run within frag
    const int l    = (tid >> 1) & 31;
    const int kt   = ((tid >> 6) & 3) + 4;            // K tiles 4..7
    const int m    = l & 15;
    const int kb   = (l & 16) ? 8 : 0;
    const int kg   = kt * 32 + kb + half * 16;        // 128..255
    const _Float16* src = h + (size_t)(row0 + m) * 128 + (kg - 128);
    _Float16* dst = As + ((size_t)kt * 32 + l) * 16 + half * 8;
    __builtin_amdgcn_global_load_async_to_lds_b128((v4i*)src, (v4i*)dst, 0, 0);
  }
  __builtin_amdgcn_s_wait_asynccnt(0);
#else
  for (int idx = 4 * 512 + tid; idx < 8 * 512; idx += 256) {
    const int e  = idx & 15;
    const int l  = (idx >> 4) & 31;
    const int kt = idx >> 9;                          // 4..7
    const int m  = l & 15;
    const int kb = (l & 16) ? 8 : 0;
    const int k  = kt * 32 + kb + frag_kk(e);         // 128..255
    As[idx] = h[(size_t)(row0 + m) * 128 + (k - 128)];
  }
#endif
  __syncthreads();

  v8f acc = {};
#pragma unroll
  for (int kt = 0; kt < 8; ++kt) {
    const v16h a = *(const v16h*)(As + ((size_t)kt * 32 + lane) * 16);
    const v16h b = *(const v16h*)(wcat + (((size_t)kt * 8 + wave) * 32 + lane) * 16);
    acc = wmma_f16(a, b, acc);
  }

  // Spill f32 tile to LDS for cross-wave LayerNorm.
  const int n     = wave * 16 + (lane & 15);
  const int rbase = (lane & 16) ? 8 : 0;
  const float bn = bias[n];
#pragma unroll
  for (int j = 0; j < 8; ++j) Os[(rbase + j) * 128 + n] = acc[j] + bn;
  __syncthreads();

  if (tid < 16) {
    float s = 0.0f, s2 = 0.0f;
#pragma unroll 4
    for (int c = 0; c < 128; ++c) { const float v = Os[tid * 128 + c]; s += v; s2 += v * v; }
    const float m   = s * (1.0f / 128.0f);
    const float var = s2 * (1.0f / 128.0f) - m * m;
    mu[tid]   = m;
    rsig[tid] = rsqrtf(var + 1e-5f);
  }
  __syncthreads();

  for (int idx = tid; idx < 16 * 128; idx += 256) {
    const int r = idx >> 7, c = idx & 127;
    float v = (Os[idx] - mu[r]) * rsig[r] * ln_g[c] + ln_b[c];
    v = fmaxf(v, 0.0f);
    h[(size_t)(row0 + r) * 128 + c] = (_Float16)v;
  }
}

// ---------------- Head: out[b] = dot(h_u[b], head_w) + head_b --------------
__global__ __launch_bounds__(256) void head_kernel(
    const _Float16* __restrict__ hu,
    const float* __restrict__ hw,
    const float* __restrict__ hb,
    float* __restrict__ out, int B)
{
  const int b = blockIdx.x * 256 + threadIdx.x;
  if (b >= B) return;
  float s = 0.0f;
#pragma unroll 4
  for (int c = 0; c < 128; ++c) s += (float)hu[(size_t)b * 128 + c] * hw[c];
  out[b] = s + hb[0];
}

// ---------------- Orchestration --------------------------------------------
extern "C" void kernel_launch(void* const* d_in, const int* in_sizes, int n_in,
                              void* d_out, int out_size, void* d_ws, size_t ws_size,
                              hipStream_t stream) {
  constexpr int NU = 100000, NI = 100000, E = 1000000, C = 128, F = 300, B = 512;

  const float* user_feat = (const float*)d_in[0];
  const float* item_feat = (const float*)d_in[1];
  const float* user_time = (const float*)d_in[2];
  const float* item_time = (const float*)d_in[3];
  const float* seed_time = (const float*)d_in[4];
  const int*   user_batch = (const int*)d_in[5];
  const int*   item_batch = (const int*)d_in[6];
  const int*   edge_ui = (const int*)d_in[7];   // [2][E] user -> item
  const int*   edge_iu = (const int*)d_in[8];   // [2][E] item -> user
  const float* enc_w_u = (const float*)d_in[9];
  const float* enc_b_u = (const float*)d_in[10];
  const float* enc_w_i = (const float*)d_in[11];
  const float* enc_b_i = (const float*)d_in[12];
  const float* time_w_u = (const float*)d_in[13];
  const float* time_b_u = (const float*)d_in[14];
  const float* time_w_i = (const float*)d_in[15];
  const float* time_b_i = (const float*)d_in[16];
  // Layer params: 17..21 u1, 22..26 i1, 27..31 u2, 32..36 i2
  const float* head_w = (const float*)d_in[37];
  const float* head_b = (const float*)d_in[38];

  // Workspace carve-up (f32 accumulators first so one zero-fill covers them).
  char* ws = (char*)d_ws;
  float* acc_u = (float*)ws;                 ws += (size_t)NU * C * 4;
  float* acc_i = (float*)ws;                 ws += (size_t)NI * C * 4;
  float* cnt_u = (float*)ws;                 ws += (size_t)NU * 4;
  float* cnt_i = (float*)ws;                 ws += (size_t)NI * 4;
  _Float16* hu = (_Float16*)ws;              ws += (size_t)NU * C * 2;
  _Float16* hi = (_Float16*)ws;              ws += (size_t)NI * C * 2;
  _Float16* wEncU = (_Float16*)ws;           ws += (size_t)320 * C * 2;
  _Float16* wEncI = (_Float16*)ws;           ws += (size_t)320 * C * 2;
  _Float16* wTimU = (_Float16*)ws;           ws += (size_t)C * C * 2;
  _Float16* wTimI = (_Float16*)ws;           ws += (size_t)C * C * 2;
  _Float16* wCat[4];                                  // u1, i1, u2, i2 (256x128)
  for (int j = 0; j < 4; ++j) { wCat[j] = (_Float16*)ws; ws += (size_t)256 * C * 2; }

  // 1) Weight conversion -> f16 fragment-major (K-pad encoders 300 -> 320).
  cvt_frag_kernel<<<(320 * C + 255) / 256, 256, 0, stream>>>(enc_w_u, wEncU, F, 320);
  cvt_frag_kernel<<<(320 * C + 255) / 256, 256, 0, stream>>>(enc_w_i, wEncI, F, 320);
  cvt_frag_kernel<<<(C * C + 255) / 256, 256, 0, stream>>>(time_w_u, wTimU, C, C);
  cvt_frag_kernel<<<(C * C + 255) / 256, 256, 0, stream>>>(time_w_i, wTimI, C, C);
  const int msgIdx[4] = {17, 22, 27, 32};             // w_msg_{u1,i1,u2,i2}
  for (int j = 0; j < 4; ++j) {
    cvt_frag2_kernel<<<(256 * C + 255) / 256, 256, 0, stream>>>(
        (const float*)d_in[msgIdx[j]], (const float*)d_in[msgIdx[j] + 1], wCat[j]);
  }

  // 2) Encoders (WMMA GEMMs + fused time encoding).
  encode_kernel<<<NU / 16, 256, 0, stream>>>(user_feat, F, user_time, seed_time,
      user_batch, wEncU, enc_b_u, wTimU, time_b_u, hu);
  encode_kernel<<<NI / 16, 256, 0, stream>>>(item_feat, F, item_time, seed_time,
      item_batch, wEncI, enc_b_i, wTimI, time_b_i, hi);

  // 3) Two SAGE layers.
  const size_t zeroN = (size_t)NU * C + (size_t)NI * C + NU + NI;
  for (int l = 0; l < 2; ++l) {
    zero_f32_kernel<<<(int)((zeroN + 255) / 256), 256, 0, stream>>>(acc_u, zeroN);
    // item -> user messages (read hi), user -> item messages (read hu)
    scatter_kernel<<<(int)(((long)E * 32 + 255) / 256), 256, 0, stream>>>(
        edge_iu, E, hi, acc_u, cnt_u);
    scatter_kernel<<<(int)(((long)E * 32 + 255) / 256), 256, 0, stream>>>(
        edge_ui, E, hu, acc_i, cnt_i);
    const int pu = (l == 0) ? 17 : 27;   // b, ln_g, ln_b at +2,+3,+4
    const int pi = (l == 0) ? 22 : 32;
    sage_update_kernel<<<NU / 16, 256, 0, stream>>>(acc_u, cnt_u, wCat[l * 2 + 0],
        (const float*)d_in[pu + 2], (const float*)d_in[pu + 3],
        (const float*)d_in[pu + 4], hu);
    sage_update_kernel<<<NI / 16, 256, 0, stream>>>(acc_i, cnt_i, wCat[l * 2 + 1],
        (const float*)d_in[pi + 2], (const float*)d_in[pi + 3],
        (const float*)d_in[pi + 4], hi);
  }

  // 4) Head on the B seed users.
  head_kernel<<<(B + 255) / 256, 256, 0, stream>>>(hu, head_w, head_b,
                                                   (float*)d_out, B);
}
